// Block_82806969467140
// MI455X (gfx1250) — matrix operand order repaired and checked by reading, not compiled, not run on hardware
//
#include <hip/hip_runtime.h>
#include <hip/hip_bf16.h>

typedef __bf16 bf16_t;
typedef __attribute__((ext_vector_type(16))) __bf16 v16bf;
typedef __attribute__((ext_vector_type(8)))  float  v8f;

union FragU { v16bf v; uint4 q[2]; };

__device__ __forceinline__ v8f wmma_bf16(v16bf a, v16bf b, v8f c) {
  // D = A(16x32 bf16) * B(32x16 bf16) + C(16x16 f32)
  return __builtin_amdgcn_wmma_f32_16x16x32_bf16(false, a, false, b, (short)0, c, false, false);
}

__device__ __forceinline__ v8f zero_v8f() {
  v8f z;
#pragma unroll
  for (int i = 0; i < 8; ++i) z[i] = 0.0f;
  return z;
}

// Async DMA: global -> LDS, 16 bytes per lane, tracked by ASYNCcnt (CDNA5).
__device__ __forceinline__ void async_copy_b128(void* lds, const void* g) {
  uint32_t l = (uint32_t)(uintptr_t)lds;  // low 32 bits of flat ptr = LDS offset
  asm volatile("global_load_async_to_lds_b128 %0, %1, off"
               :: "v"(l), "v"(g) : "memory");
}
__device__ __forceinline__ void wait_async0() {
  asm volatile("s_wait_asynccnt 0x0" ::: "memory");
}

// ---------------------------------------------------------------------------
// Weight prep: pack Wq/Wk/Wv [H,D,DH] -> bf16 [3D][D] in Bt (N-major) layout
// ---------------------------------------------------------------------------
__global__ __launch_bounds__(256) void pack_qkv_kernel(
    const float* __restrict__ Wq, const float* __restrict__ Wk,
    const float* __restrict__ Wv, bf16_t* __restrict__ out) {
  int n = blockIdx.x;            // 0..3071
  int d = threadIdx.x * 4;       // 256 threads * 4 = 1024
  const float* W = (n < 1024) ? Wq : ((n < 2048) ? Wk : Wv);
  int nn = n & 1023, h = nn >> 6, dh = nn & 63;
  bf16_t* o = out + (size_t)n * 1024 + d;
#pragma unroll
  for (int c = 0; c < 4; ++c)
    o[c] = (bf16_t)W[((size_t)h * 1024 + d + c) * 64 + dh];
}

// out[n*K + k] = W[k*N + n]  (cast f32 -> bf16, transpose to Bt layout)
__global__ __launch_bounds__(256) void transpose_cast_kernel(
    const float* __restrict__ W, bf16_t* __restrict__ out, int K, int N) {
  long idx = (long)blockIdx.x * 256 + threadIdx.x;
  if (idx >= (long)K * N) return;
  int n = (int)(idx % N);
  int k = (int)(idx / N);
  out[(size_t)n * K + k] = (bf16_t)W[idx];
}

// ---------------------------------------------------------------------------
// LayerNorm over D=1024: one block per row; writes f32 (residual) + bf16 (A op)
// ---------------------------------------------------------------------------
__global__ __launch_bounds__(256) void layernorm_kernel(
    const float* __restrict__ x, const float* __restrict__ g,
    const float* __restrict__ beta, float* __restrict__ outF,
    bf16_t* __restrict__ outB) {
  const int D = 1024;
  int row = blockIdx.x, tid = threadIdx.x;
  __shared__ float red[256];

  const float4* xr = (const float4*)(x + (size_t)row * D);
  float4 v = xr[tid];
  red[tid] = v.x + v.y + v.z + v.w;
  __syncthreads();
  for (int off = 128; off > 0; off >>= 1) {
    if (tid < off) red[tid] += red[tid + off];
    __syncthreads();
  }
  float mu = red[0] * (1.0f / D);
  __syncthreads();

  float dx0 = v.x - mu, dx1 = v.y - mu, dx2 = v.z - mu, dx3 = v.w - mu;
  red[tid] = dx0 * dx0 + dx1 * dx1 + dx2 * dx2 + dx3 * dx3;
  __syncthreads();
  for (int off = 128; off > 0; off >>= 1) {
    if (tid < off) red[tid] += red[tid + off];
    __syncthreads();
  }
  float rstd = rsqrtf(red[0] * (1.0f / D) + 1e-5f);

  float4 gv = ((const float4*)g)[tid];
  float4 bv = ((const float4*)beta)[tid];
  float y0 = dx0 * rstd * gv.x + bv.x;
  float y1 = dx1 * rstd * gv.y + bv.y;
  float y2 = dx2 * rstd * gv.z + bv.z;
  float y3 = dx3 * rstd * gv.w + bv.w;

  if (outF) {
    float4 o = make_float4(y0, y1, y2, y3);
    ((float4*)(outF + (size_t)row * D))[tid] = o;
  }
  if (outB) {
    bf16_t* ob = outB + (size_t)row * D + tid * 4;
    ob[0] = (bf16_t)y0; ob[1] = (bf16_t)y1;
    ob[2] = (bf16_t)y2; ob[3] = (bf16_t)y3;
  }
}

// ---------------------------------------------------------------------------
// Tiled bf16 WMMA GEMM: C[M,N] = A[M,K] * Bt[N,K]^T
// 128x128 block tile, 8 waves (2x4), wave tile 64x32 = 4x2 wmma accumulators.
// Double-buffered LDS staged by async global->LDS DMA; 1 barrier per K-step.
// Epilogue: optional bias[N], ReLU, residual[M,N], f32 and/or bf16 store.
// ---------------------------------------------------------------------------
__global__ __launch_bounds__(256) void gemm_bf16_wmma(
    const bf16_t* __restrict__ A, const bf16_t* __restrict__ Bt,
    int M, int N, int K,
    const float* __restrict__ bias, const float* __restrict__ residual,
    float* __restrict__ outF, bf16_t* __restrict__ outB, int relu) {
  __shared__ alignas(16) bf16_t As[2][128][32];
  __shared__ alignas(16) bf16_t Bs[2][128][32];

  int tid = threadIdx.x;
  int lane = tid & 31, w = tid >> 5;
  int wm = w >> 2, wn = w & 3;          // wave grid 2 (M) x 4 (N)
  int half = lane >> 4, l16 = lane & 15;
  int m0 = blockIdx.y * 128, n0 = blockIdx.x * 128;

  // Tile loader: thread covers rows {lr, lr+64}, 8-elem (16B) chunk lc.
  int lr = tid >> 2;
  int lc = (tid & 3) * 8;
  const bf16_t* aP0 = A + (size_t)(m0 + lr) * K + lc;
  const bf16_t* aP1 = aP0 + (size_t)64 * K;
  const bf16_t* bP0 = Bt + (size_t)(n0 + lr) * K + lc;
  const bf16_t* bP1 = bP0 + (size_t)64 * K;

  auto stage = [&](int buf, int k0) {
    async_copy_b128(&As[buf][lr][lc],      aP0 + k0);
    async_copy_b128(&As[buf][lr + 64][lc], aP1 + k0);
    async_copy_b128(&Bs[buf][lr][lc],      bP0 + k0);
    async_copy_b128(&Bs[buf][lr + 64][lc], bP1 + k0);
  };

  v8f acc[4][2];
#pragma unroll
  for (int i = 0; i < 4; ++i)
#pragma unroll
    for (int j = 0; j < 2; ++j) acc[i][j] = zero_v8f();

  stage(0, 0);  // prologue

  int nIter = K >> 5;
  for (int it = 0; it < nIter; ++it) {
    int buf = it & 1;
    wait_async0();        // my DMA into `buf` complete
    __syncthreads();      // everyone's DMA into `buf` complete; buf^1 free
    if (it + 1 < nIter) stage(buf ^ 1, (it + 1) * 32);

    FragU bq[2];
#pragma unroll
    for (int j = 0; j < 2; ++j) {
      // B frag: lane holds col = base + (lane&15); K = {half*8..+8, 16+half*8..+8}
      const bf16_t* src = &Bs[buf][wn * 32 + j * 16 + l16][0];
      bq[j].q[0] = *(const uint4*)(src + half * 8);
      bq[j].q[1] = *(const uint4*)(src + 16 + half * 8);
    }
#pragma unroll
    for (int i = 0; i < 4; ++i) {
      // A frag: lane holds row = base + (lane&15); K = {half*8..+8, 16+half*8..+8}
      FragU af;
      const bf16_t* arow = &As[buf][wm * 64 + i * 16 + l16][0];
      af.q[0] = *(const uint4*)(arow + half * 8);
      af.q[1] = *(const uint4*)(arow + 16 + half * 8);
#pragma unroll
      for (int j = 0; j < 2; ++j)
        acc[i][j] = wmma_bf16(af.v, bq[j].v, acc[i][j]);
    }
  }

  // Epilogue. C layout: VGPR r, lane -> row = r + half*8, col = lane&15.
#pragma unroll
  for (int i = 0; i < 4; ++i) {
    int rowBase = m0 + wm * 64 + i * 16 + half * 8;
#pragma unroll
    for (int j = 0; j < 2; ++j) {
      int col = n0 + wn * 32 + j * 16 + l16;
      float bval = bias ? bias[col] : 0.0f;
#pragma unroll
      for (int r = 0; r < 8; ++r) {
        int row = rowBase + r;
        float v = acc[i][j][r] + bval;
        if (relu) v = fmaxf(v, 0.0f);
        if (residual) v += residual[(size_t)row * N + col];
        if (outF) outF[(size_t)row * N + col] = v;
        if (outB) outB[(size_t)row * N + col] = (bf16_t)v;
      }
    }
  }
}

// ---------------------------------------------------------------------------
// Causal flash attention. qkv[M,3072] rows = [Q(1024) | K(1024) | V(1024)],
// head h uses columns h*64..h*64+63. One block = 128 query rows x one (b,h).
// 8 waves; wave w owns 16 query rows, full DH=64 output. Softmax in exp2 domain.
// ---------------------------------------------------------------------------
__global__ __launch_bounds__(256) void attn_kernel(
    const bf16_t* __restrict__ qkv, bf16_t* __restrict__ attnOut, int T) {
  const int RS = 3072;
  int qb = blockIdx.x * 128;
  int h = blockIdx.y;
  int b = blockIdx.z;
  int tid = threadIdx.x, lane = tid & 31, w = tid >> 5;
  int half = lane >> 4, l16 = lane & 15;

  __shared__ alignas(16) bf16_t Ks[64][64];     // K tile (row-major = Bt for QK^T)
  __shared__ alignas(16) bf16_t Vt[64][64];     // V tile transposed (Bt for PV)
  __shared__ alignas(16) bf16_t Ps[8][16][64];  // per-wave P staging (C->A relayout)

  // Q fragments held in registers for the whole kernel (A-matrix layout)
  FragU qf[2];
  {
    const bf16_t* qrow = qkv + ((size_t)b * T + qb + w * 16 + l16) * RS + h * 64;
    qf[0].q[0] = *(const uint4*)(qrow + half * 8);
    qf[0].q[1] = *(const uint4*)(qrow + 16 + half * 8);
    qf[1].q[0] = *(const uint4*)(qrow + 32 + half * 8);
    qf[1].q[1] = *(const uint4*)(qrow + 48 + half * 8);
  }

  v8f o[4];
#pragma unroll
  for (int nd = 0; nd < 4; ++nd) o[nd] = zero_v8f();
  float mrow[8], lrow[8];
#pragma unroll
  for (int r = 0; r < 8; ++r) { mrow[r] = -3.0e38f; lrow[r] = 0.0f; }
  // scores * 1/sqrt(D) in exp2 domain: scale = log2(e)/32
  const float scale2 = 1.44269504089f * 0.03125f;

  int kv = tid >> 2;            // key row 0..63
  int kc = (tid & 3) * 16;      // element offset {0,16,32,48}; covers 32B

  for (int kb = 0; kb < qb + 128; kb += 64) {   // causal: blocks <= diagonal
    __syncthreads();   // previous iteration's reads of Ks/Vt complete
    {
      const bf16_t* base = qkv + ((size_t)b * T + kb + kv) * RS + h * 64;
      // K tile via async DMA (row-major, 128B rows = chunks of 8 elems)
      async_copy_b128(&Ks[kv][kc],     base + 1024 + kc);
      async_copy_b128(&Ks[kv][kc + 8], base + 1024 + kc + 8);
      // V tile through registers, transposed into LDS
      uint4 vv0 = *(const uint4*)(base + 2048 + kc);
      uint4 vv1 = *(const uint4*)(base + 2048 + kc + 8);
      const bf16_t* ve0 = (const bf16_t*)&vv0;
      const bf16_t* ve1 = (const bf16_t*)&vv1;
#pragma unroll
      for (int e = 0; e < 8; ++e) {
        Vt[kc + e][kv]     = ve0[e];
        Vt[kc + 8 + e][kv] = ve1[e];
      }
    }
    wait_async0();
    __syncthreads();

    // S = Q * K^T : 16 x 64 per wave, 4 N-tiles, K-dim = DH = 2 wmma steps
    v8f s[4];
#pragma unroll
    for (int nt = 0; nt < 4; ++nt) {
      v8f a = zero_v8f();
#pragma unroll
      for (int kk = 0; kk < 2; ++kk) {
        FragU kf;
        const bf16_t* krow = &Ks[nt * 16 + l16][kk * 32];
        kf.q[0] = *(const uint4*)(krow + half * 8);
        kf.q[1] = *(const uint4*)(krow + 16 + half * 8);
        a = wmma_bf16(qf[kk].v, kf.v, a);
      }
      s[nt] = a;
    }

    // scale + causal mask + online softmax (row reductions over 16-lane halves)
    float mnew[8];
#pragma unroll
    for (int r = 0; r < 8; ++r) {
      int qg = qb + w * 16 + half * 8 + r;
      float mx = -3.0e38f;
#pragma unroll
      for (int nt = 0; nt < 4; ++nt) {
        int kg = kb + nt * 16 + l16;
        float v = s[nt][r] * scale2;
        if (kg > qg) v = -3.0e38f;
        s[nt][r] = v;
        mx = fmaxf(mx, v);
      }
      mx = fmaxf(mx, __shfl_xor(mx, 1));
      mx = fmaxf(mx, __shfl_xor(mx, 2));
      mx = fmaxf(mx, __shfl_xor(mx, 4));
      mx = fmaxf(mx, __shfl_xor(mx, 8));
      mnew[r] = fmaxf(mrow[r], mx);
    }
#pragma unroll
    for (int r = 0; r < 8; ++r) {
      float f = exp2f(mrow[r] - mnew[r]);
      lrow[r] *= f;
      mrow[r] = mnew[r];
#pragma unroll
      for (int nd = 0; nd < 4; ++nd) o[nd][r] *= f;
    }
#pragma unroll
    for (int r = 0; r < 8; ++r) {
      float sum = 0.0f;
#pragma unroll
      for (int nt = 0; nt < 4; ++nt) {
        float pp = exp2f(s[nt][r] - mrow[r]);
        s[nt][r] = pp;
        sum += pp;
      }
      sum += __shfl_xor(sum, 1);
      sum += __shfl_xor(sum, 2);
      sum += __shfl_xor(sum, 4);
      sum += __shfl_xor(sum, 8);
      lrow[r] += sum;
#pragma unroll
      for (int nt = 0; nt < 4; ++nt)   // C-layout -> LDS (row = r + half*8)
        Ps[w][half * 8 + r][nt * 16 + l16] = (bf16_t)s[nt][r];
    }
    asm volatile("s_wait_dscnt 0x0" ::: "memory");  // wave-local LDS RAW fence

    // O += P * V : A frags from Ps, B frags from transposed V tile
    FragU pf[2];
#pragma unroll
    for (int kk = 0; kk < 2; ++kk) {
      const bf16_t* prow = &Ps[w][l16][kk * 32];
      pf[kk].q[0] = *(const uint4*)(prow + half * 8);
      pf[kk].q[1] = *(const uint4*)(prow + 16 + half * 8);
    }
#pragma unroll
    for (int nd = 0; nd < 4; ++nd) {
#pragma unroll
      for (int kk = 0; kk < 2; ++kk) {
        FragU vf;
        const bf16_t* vrow = &Vt[nd * 16 + l16][kk * 32];
        vf.q[0] = *(const uint4*)(vrow + half * 8);
        vf.q[1] = *(const uint4*)(vrow + 16 + half * 8);
        o[nd] = wmma_bf16(pf[kk].v, vf.v, o[nd]);
      }
    }
  }

  // finalize: O /= l, write concat-head layout [b,t, h*64+dh]
#pragma unroll
  for (int r = 0; r < 8; ++r) {
    int qg = qb + w * 16 + half * 8 + r;
    float inv = 1.0f / lrow[r];
    bf16_t* orow = attnOut + ((size_t)b * T + qg) * 1024 + h * 64;
#pragma unroll
    for (int nd = 0; nd < 4; ++nd)
      orow[nd * 16 + l16] = (bf16_t)(o[nd][r] * inv);
  }
}

// ---------------------------------------------------------------------------
extern "C" void kernel_launch(void* const* d_in, const int* in_sizes, int n_in,
                              void* d_out, int out_size, void* d_ws, size_t ws_size,
                              hipStream_t stream) {
  const float* x   = (const float*)d_in[0];
  // d_in[1] = eos_mask (tril) -> implemented analytically as causal mask
  const float* Wq  = (const float*)d_in[2];
  const float* Wk  = (const float*)d_in[3];
  const float* Wv  = (const float*)d_in[4];
  const float* Wp  = (const float*)d_in[5];
  const float* bp  = (const float*)d_in[6];
  const float* W1  = (const float*)d_in[7];
  const float* b1  = (const float*)d_in[8];
  const float* W2  = (const float*)d_in[9];
  const float* b2  = (const float*)d_in[10];
  const float* g1  = (const float*)d_in[11];
  const float* be1 = (const float*)d_in[12];
  const float* g2  = (const float*)d_in[13];
  const float* be2 = (const float*)d_in[14];

  const int Bb = 4, T = 2048, D = 1024, H = 16;
  const int M = Bb * T, DF = 4 * D, N3 = 3 * D;

  char* p = (char*)d_ws;
  auto alloc = [&](size_t bytes) {
    char* r = p;
    p += (bytes + 255) & ~(size_t)255;
    return r;
  };
  float*  x1F   = (float*) alloc((size_t)M * D * 4);
  bf16_t* x1B   = (bf16_t*)alloc((size_t)M * D * 2);
  bf16_t* wqkv  = (bf16_t*)alloc((size_t)N3 * D * 2);
  bf16_t* wpt   = (bf16_t*)alloc((size_t)D * D * 2);
  bf16_t* w1t   = (bf16_t*)alloc((size_t)DF * D * 2);
  bf16_t* w2t   = (bf16_t*)alloc((size_t)D * DF * 2);
  bf16_t* qkv   = (bf16_t*)alloc((size_t)M * N3 * 2);
  bf16_t* attnB = (bf16_t*)alloc((size_t)M * D * 2);
  float*  x2F   = (float*) alloc((size_t)M * D * 4);
  float*  x3F   = (float*) alloc((size_t)M * D * 4);
  bf16_t* x3B   = (bf16_t*)alloc((size_t)M * D * 2);
  bf16_t* hB    = (bf16_t*)alloc((size_t)M * DF * 2);

  // weight prep (bf16 cast + transpose to Bt layout)
  pack_qkv_kernel<<<N3, 256, 0, stream>>>(Wq, Wk, Wv, wqkv);
  transpose_cast_kernel<<<(D * D + 255) / 256, 256, 0, stream>>>(Wp, wpt, D, D);
  transpose_cast_kernel<<<(D * DF + 255) / 256, 256, 0, stream>>>(W1, w1t, D, DF);
  transpose_cast_kernel<<<(DF * D + 255) / 256, 256, 0, stream>>>(W2, w2t, DF, D);

  // x1 = LN(x)
  layernorm_kernel<<<M, 256, 0, stream>>>(x, g1, be1, x1F, x1B);
  // fused QKV projection: [M,3072] bf16
  gemm_bf16_wmma<<<dim3(N3 / 128, M / 128), 256, 0, stream>>>(
      x1B, wqkv, M, N3, D, nullptr, nullptr, nullptr, qkv, 0);
  // causal flash attention -> concat heads [M,1024] bf16
  attn_kernel<<<dim3(T / 128, H, Bb), 256, 0, stream>>>(qkv, attnB, T);
  // x2 = x1 + attn @ Wp + bp
  gemm_bf16_wmma<<<dim3(D / 128, M / 128), 256, 0, stream>>>(
      attnB, wpt, M, D, D, bp, x1F, x2F, nullptr, 0);
  // x3 = LN(x2)
  layernorm_kernel<<<M, 256, 0, stream>>>(x2F, g2, be2, x3F, x3B);
  // h = relu(x3 @ W1 + b1)
  gemm_bf16_wmma<<<dim3(DF / 128, M / 128), 256, 0, stream>>>(
      x3B, w1t, M, DF, D, b1, nullptr, nullptr, hB, 1);
  // out = x3 + h @ W2 + b2
  gemm_bf16_wmma<<<dim3(D / 128, M / 128), 256, 0, stream>>>(
      hB, w2t, M, D, DF, b2, x3F, (float*)d_out, nullptr, 0);
}
